// Model_39676907882675
// MI455X (gfx1250) — matrix operand order repaired and checked by reading, not compiled, not run on hardware
//
#include <hip/hip_runtime.h>
#include <hip/hip_bf16.h>

// ---------------------------------------------------------------------------
// Attention: out = V @ softmax(Q @ K^T)   (B=8, S=D=2048, fp32 in/out)
// bf16 WMMA (v_wmma_f32_16x16x32_bf16) for both GEMMs, f32 accumulation.
// GEMM2's bf16 B-operand is streamed with global_load_async_to_lds_b128
// (ASYNCcnt-tracked, LDS double buffer).
// Workspace layout (needs ~201.5 MB):
//   [0)                qk   : B*S*S f32   (134 MB)  -- mostly L2-resident (192MB L2)
//   [B*S*S*4)          Pt   : B*S*S bf16  ( 67 MB)  -- softmax(qk) TRANSPOSED
//   [B*S*S*6)          rmax : B*S  f32
//   [.. + B*S*4)       rinv : B*S  f32
// ---------------------------------------------------------------------------

#define BATCH 8
#define SEQ   2048

typedef __attribute__((ext_vector_type(16))) __bf16 v16bf;
typedef __attribute__((ext_vector_type(8)))  __bf16 v8bf;
typedef __attribute__((ext_vector_type(8)))  float  v8f;
typedef __attribute__((ext_vector_type(16))) float  v16f;

#define BM 256
#define BN 128
#define BK 32
#define LDSTR 40            // padded LDS row stride (bf16 elems) -> conflict-free frags
#define BBUF  (BN * LDSTR)  // one B buffer, in bf16 elems

union Frag { v16bf v; v8bf h[2]; };

// Load 16 contiguous f32, convert to bf16 (v_cvt_pk_bf16_f32, RTNE).
__device__ __forceinline__ v16bf load16(const float* p) {
  union { v16f f; float4 q[4]; } u;
  const float4* p4 = (const float4*)p;
  u.q[0] = p4[0]; u.q[1] = p4[1]; u.q[2] = p4[2]; u.q[3] = p4[3];
  return __builtin_convertvector(u.f, v16bf);
}
// Load 16 contiguous bf16.
__device__ __forceinline__ v16bf load16(const __bf16* p) {
  Frag u;
  u.h[0] = *(const v8bf*)p;
  u.h[1] = *(const v8bf*)(p + 8);
  return u.v;
}

// Async DMA of one 16-byte chunk: global -> LDS (per-lane addresses).
__device__ __forceinline__ void async_b128(unsigned lds_off, const void* g) {
  asm volatile("global_load_async_to_lds_b128 %0, %1, off"
               :: "v"(lds_off), "v"((unsigned long long)(uintptr_t)g)
               : "memory");
}

// C[m,n] = sum_k A[m,k] * Bm[n,k]   ("NT": both operands row-major along k)
// A is f32 (converted on the fly). Bm f32 -> register-staged; bf16 -> async DMA.
template <typename TB>
__global__ __launch_bounds__(256) void gemm_nt_bf16(const float* __restrict__ A,
                                                    const TB* __restrict__ Bm,
                                                    float* __restrict__ C) {
  constexpr bool ASYNC_B = (sizeof(TB) == 2);
  __shared__ __align__(16) __bf16 smem[BM * LDSTR + (ASYNC_B ? 2 : 1) * BBUF];
  __bf16* const As = smem;
  __bf16* const Bs = smem + BM * LDSTR;

  const int t  = threadIdx.x;
  const int m0 = blockIdx.y * BM;
  const int n0 = blockIdx.x * BN;
  const size_t bofs = (size_t)blockIdx.z * SEQ * SEQ;

  // ---- A staging: 256x32 f32 tile, two 16-elem chunks per thread
  const int arow = t >> 1;          // 0..127
  const int acol = (t & 1) * 16;    // 0 or 16
  const float* Ag0 = A + bofs + (size_t)(m0 + arow) * SEQ + acol;
  const float* Ag1 = Ag0 + (size_t)128 * SEQ;
  __bf16* sa0 = As + arow * LDSTR + acol;
  __bf16* sa1 = sa0 + 128 * LDSTR;

  // ---- B staged path (f32 source): 128x32 tile, one chunk per thread
  const TB* Bg = Bm + bofs + (size_t)(n0 + arow) * SEQ + acol;
  __bf16* sb = Bs + arow * LDSTR + acol;

  // ---- B async path (bf16 source): 512 x 16B chunks, 2 per thread
  const int br0  = t >> 2;          // rows 0..63
  const int br1  = br0 + 64;        // rows 64..127
  const int bsub = (t & 3) * 16;    // byte offset of 16B chunk within a 64B row
  const char* BgaBase = (const char*)(Bm + bofs);
  const char* Bga0 = BgaBase + (size_t)(n0 + br0) * SEQ * 2 + bsub;
  const char* Bga1 = BgaBase + (size_t)(n0 + br1) * SEQ * 2 + bsub;
  // LDS byte addresses: low 32 bits of the flat shared pointer are the LDS offset
  const unsigned lb0 = (unsigned)(uintptr_t)(Bs + br0 * LDSTR) + bsub;
  const unsigned lb1 = (unsigned)(uintptr_t)(Bs + br1 * LDSTR) + bsub;
  auto issueB = [&](int kt, int buf) {
    const unsigned bo = (unsigned)buf * (BBUF * 2);
    async_b128(lb0 + bo, Bga0 + (size_t)kt * 2);
    async_b128(lb1 + bo, Bga1 + (size_t)kt * 2);
  };

  // ---- Wave tiling: 8 waves -> 4 (M) x 2 (N); wave tile 64x64 = 4x4 WMMA frags
  const int lane = t & 31;
  const int wave = t >> 5;
  const int wm0  = (wave & 3) * 64;
  const int wn0  = (wave >> 2) * 64;
  const int half = lane >> 4;
  const int rr   = lane & 15;

  const v8f vz = {0.f,0.f,0.f,0.f,0.f,0.f,0.f,0.f};
  v8f acc[4][4];
#pragma unroll
  for (int fm = 0; fm < 4; ++fm)
#pragma unroll
    for (int fn = 0; fn < 4; ++fn) acc[fm][fn] = vz;

  Frag a0r, a1r, brg;
  a0r.v = load16(Ag0);
  a1r.v = load16(Ag1);
  if constexpr (ASYNC_B) issueB(0, 0);
  else                   brg.v = load16(Bg);

  int ib = 0;
  for (int kt = 0; kt < SEQ; kt += BK, ib ^= 1) {
    __syncthreads();                             // prior compute done reading LDS
    *(v8bf*)sa0 = a0r.h[0]; *(v8bf*)(sa0 + 8) = a0r.h[1];
    *(v8bf*)sa1 = a1r.h[0]; *(v8bf*)(sa1 + 8) = a1r.h[1];

    const int kn = (kt + BK < SEQ) ? kt + BK : kt;
    if constexpr (ASYNC_B) {
      issueB(kn, ib ^ 1);                        // DMA next tile into other buffer
    } else {
      *(v8bf*)sb = brg.h[0]; *(v8bf*)(sb + 8) = brg.h[1];
    }

    a0r.v = load16(Ag0 + kn);                    // stage next A while computing
    a1r.v = load16(Ag1 + kn);
    if constexpr (!ASYNC_B) brg.v = load16(Bg + kn);
    if (kt + 2 * BK < SEQ) {                     // global_prefetch_b8
      __builtin_prefetch(Ag0 + kt + 2 * BK, 0, 0);
      __builtin_prefetch(Ag1 + kt + 2 * BK, 0, 0);
    }

    if constexpr (ASYNC_B) {
      // 2 chunks/tile/wave, in-order completion: <=2 outstanding retires the
      // pair issued last iteration (i.e. the buffer we are about to read).
      asm volatile("s_wait_asynccnt 0x2" ::: "memory");
    }
    __syncthreads();                             // tile (A stores + B DMA) visible

    const __bf16* Bcur = Bs + (ASYNC_B ? ib * BBUF : 0);
    // Fragment gather per CDNA5 16-bit operand layouts (CDNA5 ISA §7.12.2)
    Frag a[4], b[4];
#pragma unroll
    for (int fm = 0; fm < 4; ++fm) {
      const __bf16* pa = &As[(wm0 + fm * 16 + rr) * LDSTR];
      a[fm].h[0] = *(const v8bf*)(pa + half * 8);
      a[fm].h[1] = *(const v8bf*)(pa + 16 + half * 8);
    }
#pragma unroll
    for (int fn = 0; fn < 4; ++fn) {
      const __bf16* pb = &Bcur[(wn0 + fn * 16 + rr) * LDSTR + half * 8];
      b[fn].h[0] = *(const v8bf*)pb;
      b[fn].h[1] = *(const v8bf*)(pb + 16);
    }
#pragma unroll
    for (int fm = 0; fm < 4; ++fm)
#pragma unroll
      for (int fn = 0; fn < 4; ++fn)
        acc[fm][fn] = __builtin_amdgcn_wmma_f32_16x16x32_bf16(
            false, a[fm].v, false, b[fn].v, (short)0, acc[fm][fn], false, false);
  }

  // C/D layout: VGPR i -> M = i + 8*(lane>>4), N = lane&15
  const int cr = half * 8;
  float* Cb = C + bofs;
#pragma unroll
  for (int fm = 0; fm < 4; ++fm)
#pragma unroll
    for (int fn = 0; fn < 4; ++fn) {
      float* cp = Cb + (size_t)(m0 + wm0 + fm * 16 + cr) * SEQ + (n0 + wn0 + fn * 16 + rr);
#pragma unroll
      for (int i = 0; i < 8; ++i) cp[(size_t)i * SEQ] = acc[fm][fn][i];
    }
}

// Per-row max and reciprocal sum-of-exp (qk ~ N(0,2048): max-subtraction required).
__global__ __launch_bounds__(256) void row_stats(const float* __restrict__ qk,
                                                 float* __restrict__ rmax,
                                                 float* __restrict__ rinv) {
  __shared__ float red[256];
  const int t = threadIdx.x;
  const size_t row = (size_t)blockIdx.y * SEQ + blockIdx.x;
  const float* p = qk + row * SEQ;
  float v[8];
#pragma unroll
  for (int i = 0; i < 8; ++i) v[i] = p[t + i * 256];
  float m = v[0];
#pragma unroll
  for (int i = 1; i < 8; ++i) m = fmaxf(m, v[i]);
  red[t] = m; __syncthreads();
  for (int o = 128; o > 0; o >>= 1) {
    if (t < o) red[t] = fmaxf(red[t], red[t + o]);
    __syncthreads();
  }
  m = red[0]; __syncthreads();
  float s = 0.f;
#pragma unroll
  for (int i = 0; i < 8; ++i) s += __expf(v[i] - m);
  red[t] = s; __syncthreads();
  for (int o = 128; o > 0; o >>= 1) {
    if (t < o) red[t] += red[t + o];
    __syncthreads();
  }
  if (t == 0) { rmax[row] = m; rinv[row] = 1.0f / red[0]; }
}

// Pt[b, j, i] = exp(qk[b, i, j] - rmax[b,i]) * rinv[b,i]  (bf16, LDS transpose)
#define TT   64
#define TSTR 72
__global__ __launch_bounds__(256) void softmax_transpose(const float* __restrict__ qk,
                                                         const float* __restrict__ rmax,
                                                         const float* __restrict__ rinv,
                                                         __bf16* __restrict__ Pt) {
  __shared__ __align__(16) __bf16 tr[TT * TSTR];  // tr[j_local * TSTR + i_local]
  const int i0 = blockIdx.x * TT;
  const int j0 = blockIdx.y * TT;
  const size_t bq = (size_t)blockIdx.z * SEQ * SEQ;
  const int t = threadIdx.x;
  const int rbase = t >> 4;
  const int c4 = (t & 15) * 4;
#pragma unroll
  for (int it = 0; it < 4; ++it) {
    const int r = rbase + it * 16;
    const float4 v = *(const float4*)(qk + bq + (size_t)(i0 + r) * SEQ + j0 + c4);
    const size_t grow = (size_t)blockIdx.z * SEQ + i0 + r;
    const float mx = rmax[grow];
    const float ri = rinv[grow];
    tr[(c4 + 0) * TSTR + r] = (__bf16)(__expf(v.x - mx) * ri);
    tr[(c4 + 1) * TSTR + r] = (__bf16)(__expf(v.y - mx) * ri);
    tr[(c4 + 2) * TSTR + r] = (__bf16)(__expf(v.z - mx) * ri);
    tr[(c4 + 3) * TSTR + r] = (__bf16)(__expf(v.w - mx) * ri);
  }
  __syncthreads();
  const int j   = t >> 2;
  const int i16 = (t & 3) * 16;
  Frag u;
  u.h[0] = *(const v8bf*)&tr[j * TSTR + i16];
  u.h[1] = *(const v8bf*)&tr[j * TSTR + i16 + 8];
  __bf16* dst = Pt + bq + (size_t)(j0 + j) * SEQ + i0 + i16;
  *(v8bf*)dst = u.h[0];
  *(v8bf*)(dst + 8) = u.h[1];
}

extern "C" void kernel_launch(void* const* d_in, const int* in_sizes, int n_in,
                              void* d_out, int out_size, void* d_ws, size_t ws_size,
                              hipStream_t stream) {
  const float* Q  = (const float*)d_in[0];
  const float* Kv = (const float*)d_in[1];
  const float* V  = (const float*)d_in[2];
  (void)d_in[3];  // attn_mask: all zeros, unused by the reference math
  (void)in_sizes; (void)n_in; (void)out_size; (void)ws_size;

  char* ws = (char*)d_ws;
  float*  qk   = (float*)ws;                                    // B*S*S f32
  __bf16* Pt   = (__bf16*)(ws + (size_t)BATCH * SEQ * SEQ * 4); // B*S*S bf16
  float*  rmax = (float*)(ws + (size_t)BATCH * SEQ * SEQ * 6);
  float*  rinv = rmax + (size_t)BATCH * SEQ;

  const dim3 blk(256);
  const dim3 gg(SEQ / BN, SEQ / BM, BATCH);

  // 1) qk = Q @ K^T           (bf16 WMMA, f32 accum -> ws)
  gemm_nt_bf16<float><<<gg, blk, 0, stream>>>(Q, Kv, qk);
  // 2) per-row softmax stats
  row_stats<<<dim3(SEQ, BATCH), blk, 0, stream>>>(qk, rmax, rinv);
  // 3) Pt = softmax(qk)^T     (bf16, transposed so GEMM2's reduction dim is contiguous)
  softmax_transpose<<<dim3(SEQ / TT, SEQ / TT, BATCH), blk, 0, stream>>>(qk, rmax, rinv, Pt);
  // 4) out = V @ P = V @ Pt^T (bf16 WMMA async-B path, f32 accum -> d_out)
  gemm_nt_bf16<__bf16><<<gg, blk, 0, stream>>>(V, Pt, (float*)d_out);
}
